// L4QQuantizedLinear_80745385165259
// MI455X (gfx1250) — matrix operand order repaired and checked by reading, not compiled, not run on hardware
//
#include <hip/hip_runtime.h>
#include <hip/hip_bf16.h>

// ---------------------------------------------------------------------------
// L4Q quantized linear for MI455X (gfx1250, wave32, WMMA)
//   w_q = fake_quant(w0 + lora_b@lora_a), out = x @ w_q^T + bias
//   M=1024, N=4096, K=4096, RANK=16, GROUP=128
// Strategy: pre-swizzle x (bf16) and w_q (bf16) into per-lane WMMA fragment
// order in workspace; GEMM is pure global_load_b128 + v_wmma_f32_16x16x32_bf16.
// Operands (8MB + 32MB) are L2-resident (192MB), so no LDS staging is needed.
// ---------------------------------------------------------------------------

typedef __attribute__((ext_vector_type(16))) __bf16 v16bf;
typedef __attribute__((ext_vector_type(8)))  float  v8f;

#define IN_F   4096
#define OUT_F  4096
#define M_TOT  1024
#define RANK   16
#define KT_CNT (IN_F / 32)          // 128 k-tiles of 32
#define FRAG_BYTES 1024             // 32 lanes * 32 bytes (16 bf16) per fragment

__device__ __forceinline__ unsigned f2bf_rne(float f) {
    unsigned u = __builtin_bit_cast(unsigned, f);
    u += 0x7FFFu + ((u >> 16) & 1u);      // round-to-nearest-even
    return u >> 16;
}

// ---------------------------------------------------------------------------
// Kernel 1: w_q = fake_quant(w0 + lora_b@lora_a) -> bf16, scattered into
// B-fragment layout: frag index = n_tile*KT_CNT + k_tile, within frag:
//   lane = col + 16*((k>>3)&1), vgpr = ((k&7)>>1) + 4*(k>>4)
//   (halfword = k&1)
// Each thread handles 2 consecutive k (one packed b32 write).
// Grid: OUT_F * 8 blocks of 256 threads (o is uniform per block -> s_loads).
// ---------------------------------------------------------------------------
__global__ __launch_bounds__(256)
void prep_w(const float* __restrict__ w0, const float* __restrict__ la,
            const float* __restrict__ lb, const float* __restrict__ qs,
            unsigned* __restrict__ bfrag) {
    const int o  = blockIdx.x >> 3;                       // output row, uniform
    const int ip = ((blockIdx.x & 7) << 8) + threadIdx.x; // pair index 0..2047
    const int i  = ip << 1;                               // even column

    float2 w = *(const float2*)(w0 + (size_t)o * IN_F + i);
    float aL = w.x, aR = w.y;
#pragma unroll
    for (int r = 0; r < RANK; ++r) {
        const float b = lb[o * RANK + r];                 // block-uniform
        const float2 av = *(const float2*)(la + r * IN_F + i);
        aL += b * av.x;
        aR += b * av.y;
    }
    // fake-quant, group = o*32 + i/128 (GROUP=128 divides IN_F)
    const float s   = qs[(o << 5) + (i >> 7)];
    const float inv = 1.0f / (s + 1e-9f);
    float qL = __builtin_rintf(fminf(7.0f, fmaxf(-8.0f, aL * inv)));
    float qR = __builtin_rintf(fminf(7.0f, fmaxf(-8.0f, aR * inv)));
    const unsigned pack = f2bf_rne(qL * s) | (f2bf_rne(qR * s) << 16);

    const int col = o & 15, nt = o >> 4;
    const int k = i & 31, kt = i >> 5;
    const int lane = col + (((k >> 3) & 1) << 4);
    const int vg   = ((k & 7) >> 1) + ((k >> 4) << 2);
    bfrag[((nt * KT_CNT + kt) << 8) + lane * 8 + vg] = pack;
}

// ---------------------------------------------------------------------------
// Kernel 2: x f32 -> bf16 in A-fragment layout (16x32 MxK tiles).
//   kk<8  : lane=row,    vg=kk/2        kk in [8,16) : lane=row+16, vg=(kk-8)/2
//   kk<24 : lane=row,    vg=4+(kk-16)/2 kk>=24       : lane=row+16, vg=4+(kk-24)/2
// ---------------------------------------------------------------------------
__global__ __launch_bounds__(256)
void conv_x(const float* __restrict__ x, unsigned* __restrict__ afrag) {
    const int m  = blockIdx.x >> 3;                       // row 0..1023, uniform
    const int ip = ((blockIdx.x & 7) << 8) + threadIdx.x; // pair 0..2047
    const int kg = ip << 1;

    float2 v = *(const float2*)(x + (size_t)m * IN_F + kg);
    const unsigned pack = f2bf_rne(v.x) | (f2bf_rne(v.y) << 16);

    const int row = m & 15, mt = m >> 4;
    const int kk = kg & 31, kt = kg >> 5;
    int lane, vg;
    if (kk < 8)       { lane = row;      vg = kk >> 1; }
    else if (kk < 16) { lane = row + 16; vg = (kk - 8) >> 1; }
    else if (kk < 24) { lane = row;      vg = 4 + ((kk - 16) >> 1); }
    else              { lane = row + 16; vg = 4 + ((kk - 24) >> 1); }
    afrag[((mt * KT_CNT + kt) << 8) + lane * 8 + vg] = pack;
}

// ---------------------------------------------------------------------------
// Kernel 3: GEMM via v_wmma_f32_16x16x32_bf16.
// 256 threads = 8 waves; block tile 64(M) x 256(N); wave tile 32 x 64
//   -> 2 (M) x 4 (N) accumulators, 8 WMMAs per K-step, 128 K-steps.
// C layout: N = lane&15, M = vgpr_r + 8*(lane>=16).
// ---------------------------------------------------------------------------
__global__ __launch_bounds__(256)
void gemm_wmma(const char* __restrict__ aBytes, const char* __restrict__ bBytes,
               const float* __restrict__ bias, float* __restrict__ out) {
    const int lane   = threadIdx.x & 31;
    const int wave   = threadIdx.x >> 5;
    const int wave_m = wave & 1;          // 0..1
    const int wave_n = wave >> 1;         // 0..3

    const int m_wave = blockIdx.y * 64  + wave_m * 32;   // wave's first row
    const int n_wave = blockIdx.x * 256 + wave_n * 64;   // wave's first col
    const int laneOff = lane * 32;

    const char* aB0 = aBytes + (size_t)(((m_wave >> 4) + 0) * KT_CNT) * FRAG_BYTES + laneOff;
    const char* aB1 = aBytes + (size_t)(((m_wave >> 4) + 1) * KT_CNT) * FRAG_BYTES + laneOff;
    const char* bB[4];
#pragma unroll
    for (int j = 0; j < 4; ++j)
        bB[j] = bBytes + (size_t)(((n_wave >> 4) + j) * KT_CNT) * FRAG_BYTES + laneOff;

    v8f acc0[4], acc1[4];
#pragma unroll
    for (int j = 0; j < 4; ++j) { acc0[j] = (v8f)0.0f; acc1[j] = (v8f)0.0f; }

    for (int kt = 0; kt < KT_CNT; ++kt) {
        const int koff = kt << 10;
        // prefetch a few tiles ahead (global_prefetch_b8)
        __builtin_prefetch(aB0 + koff + 4 * FRAG_BYTES, 0, 1);
        __builtin_prefetch(bB[0] + koff + 4 * FRAG_BYTES, 0, 1);

        v16bf a0 = *(const v16bf*)(aB0 + koff);
        v16bf a1 = *(const v16bf*)(aB1 + koff);
#pragma unroll
        for (int j = 0; j < 4; ++j) {
            v16bf b = *(const v16bf*)(bB[j] + koff);
            acc0[j] = __builtin_amdgcn_wmma_f32_16x16x32_bf16(
                false, a0, false, b, (short)0, acc0[j], false, false);
            acc1[j] = __builtin_amdgcn_wmma_f32_16x16x32_bf16(
                false, a1, false, b, (short)0, acc1[j], false, false);
        }
    }

    // epilogue: + bias, store f32
    const int colBase = n_wave + (lane & 15);
    const int rowBase = m_wave + ((lane >> 4) << 3);      // +8 for lanes 16..31
#pragma unroll
    for (int j = 0; j < 4; ++j) {
        const int col = colBase + j * 16;
        const float bj = bias[col];
#pragma unroll
        for (int r = 0; r < 8; ++r) {
            out[(size_t)(rowBase + r)      * OUT_F + col] = acc0[j][r] + bj;
            out[(size_t)(rowBase + 16 + r) * OUT_F + col] = acc1[j][r] + bj;
        }
    }
}

// ---------------------------------------------------------------------------
extern "C" void kernel_launch(void* const* d_in, const int* in_sizes, int n_in,
                              void* d_out, int out_size, void* d_ws, size_t ws_size,
                              hipStream_t stream) {
    const float* x    = (const float*)d_in[0];
    const float* w0   = (const float*)d_in[1];
    const float* la   = (const float*)d_in[2];
    const float* lb   = (const float*)d_in[3];
    const float* qs   = (const float*)d_in[4];
    const float* bias = (const float*)d_in[5];
    float* out = (float*)d_out;

    // workspace layout: [0, 8MB) A-fragments (x bf16), [8MB, 40MB) B-fragments
    char* ws = (char*)d_ws;
    unsigned* afrag = (unsigned*)ws;                                   // 8 MB
    unsigned* bfrag = (unsigned*)(ws + (size_t)M_TOT * IN_F * 2);      // 32 MB

    prep_w<<<OUT_F * 8, 256, 0, stream>>>(w0, la, lb, qs, bfrag);
    conv_x<<<M_TOT * 8, 256, 0, stream>>>(x, afrag);

    dim3 grid(OUT_F / 256, M_TOT / 64);   // 16 x 16 blocks
    gemm_wmma<<<grid, 256, 0, stream>>>((const char*)afrag, (const char*)bfrag,
                                        bias, out);
}
